// PointerNetwork_63393717289581
// MI455X (gfx1250) — compile-verified, matrix operand order checked by reading, not run on hardware
//
#include <hip/hip_runtime.h>

// ---------------------------------------------------------------------------
// Pointer network decoder for MI455X (gfx1250, wave32, WMMA).
// Strategy:
//   * Precompute e_g/e_p = Wref @ context^T + b once (WMMA bf16), store bf16
//     (32MB each) so both stay resident in the 192MB L2 across all 126 steps.
//   * Per step: one fused-gate WMMA GEMM (gates = [x||h] @ [Wih||Whh]^T + b),
//     elementwise LSTM, WMMA GEMM qg, attention/glimpse kernel, WMMA GEMM qp,
//     attention/pointer kernel (softmax -> probs out, argmax -> mask update +
//     gather of next embedded input).
// ---------------------------------------------------------------------------

typedef __attribute__((ext_vector_type(8)))  __bf16 v8bf;
typedef __attribute__((ext_vector_type(16))) __bf16 v16bf;
typedef __attribute__((ext_vector_type(8)))  float  v8f;

constexpr int B_    = 2048;
constexpr int L_    = 64;
constexpr int E_    = 128;
constexpr int H_    = 128;
constexpr int G4_   = 512;   // 4*H
constexpr int KCAT  = 256;   // E + H
constexpr int STEPS = 126;   // (L-1)*2
constexpr float NEGC  = -1e9f;
constexpr float CTANH = 10.0f;

#define DEV static __device__ __forceinline__

// --- bf16 helpers (storage-only use of __bf16; explicit RNE convert) --------
DEV __bf16 f2bf(float f) {
    union { float f; unsigned u; } v; v.f = f;
    unsigned r = v.u + 0x7FFFu + ((v.u >> 16) & 1u);   // round-to-nearest-even
    unsigned short h = (unsigned short)(r >> 16);
    return __builtin_bit_cast(__bf16, h);
}
DEV float bf2f(__bf16 b) {
    unsigned short h = __builtin_bit_cast(unsigned short, b);
    union { unsigned u; float f; } v; v.u = ((unsigned)h) << 16;
    return v.f;
}

// Load one wave-lane's 16 bf16 elements of a 16x32 WMMA operand tile.
// ISA layout: lanes 0-15 hold K = {k0..k0+7, k0+16..k0+23};
//             lanes 16-31 hold K = {k0+8..k0+15, k0+24..k0+31}.
// `p` must already point at  row_base + k0 + half*8.
union V16U { v16bf v; struct { v8bf lo; v8bf hi; } p; };
DEV v16bf load_frag16(const __bf16* p) {
    V16U u;
    u.p.lo = *(const v8bf*)(p);        // K = half*8 .. +7
    u.p.hi = *(const v8bf*)(p + 16);   // K = 16 + half*8 .. +7
    return u.v;
}

// --- generic bf16 WMMA GEMM:  out(MxN,f32) = A(MxK) @ W(NxK)^T + bias(N) ----
// one 16x16 tile per wave; grid*8 waves must equal (M/16)*(N/16)
__global__ void k_gemm_bias(const __bf16* __restrict__ A, int lda,
                            const __bf16* __restrict__ W, int K,
                            const float* __restrict__ bias,
                            float* __restrict__ out, int ldo,
                            int tilesN)
{
    const int wave = blockIdx.x * (blockDim.x >> 5) + (threadIdx.x >> 5);
    const int lane = threadIdx.x & 31;
    const int half = lane >> 4;
    const int tn = wave % tilesN;
    const int tm = wave / tilesN;
    const int mrow = tm * 16 + (lane & 15);
    const int ncol = tn * 16 + (lane & 15);

    const __bf16* pa = A + (size_t)mrow * lda + half * 8;
    const __bf16* pw = W + (size_t)ncol * K   + half * 8;

    v8f acc = {};
    for (int k = 0; k < K; k += 32) {
        v16bf a = load_frag16(pa + k);
        v16bf w = load_frag16(pw + k);
        acc = __builtin_amdgcn_wmma_f32_16x16x32_bf16(
                  false, a, false, w, (short)0, acc, false, false);
    }
    const float bb = bias[ncol];
    float* po = out + (size_t)(tm * 16 + half * 8) * ldo + ncol;
#pragma unroll
    for (int i = 0; i < 8; ++i) po[(size_t)i * ldo] = acc[i] + bb;
}

// --- e-tensor precompute: e[b,h,l] = sum_k Wref[h,k]*ctx[l,b,k] + bref[h] ---
// A rows are the (l*B+b) flattening of context (bf16), tilesN = H/16 = 8.
__global__ void k_eref(const __bf16* __restrict__ ctx,   // (L*B) x H bf16
                       const __bf16* __restrict__ Wref,  // H x H bf16
                       const float*  __restrict__ bref,  // H f32
                       __bf16* __restrict__ e)           // (B,H,L) bf16
{
    const int wave = blockIdx.x * (blockDim.x >> 5) + (threadIdx.x >> 5);
    const int lane = threadIdx.x & 31;
    const int half = lane >> 4;
    const int tn = wave & 7;
    const int tm = wave >> 3;
    const int r0 = tm * 16;              // flattened row = l*B + b
    const int l  = r0 >> 11;             // / B_
    const int b0 = r0 & (B_ - 1);        // % B_   (16 | B_, tiles never cross l)
    const int mrow = r0 + (lane & 15);
    const int ncol = tn * 16 + (lane & 15);

    const __bf16* pa = ctx  + (size_t)mrow * H_ + half * 8;
    const __bf16* pw = Wref + (size_t)ncol * H_ + half * 8;

    v8f acc = {};
    for (int k = 0; k < H_; k += 32) {
        v16bf a = load_frag16(pa + k);
        v16bf w = load_frag16(pw + k);
        acc = __builtin_amdgcn_wmma_f32_16x16x32_bf16(
                  false, a, false, w, (short)0, acc, false, false);
    }
    const float bb = bref[ncol];
#pragma unroll
    for (int i = 0; i < 8; ++i) {
        const int brow = b0 + half * 8 + i;
        e[(size_t)brow * (H_ * L_) + (size_t)ncol * L_ + l] = f2bf(acc[i] + bb);
    }
}

// --- f32 -> bf16 convert (grid-stride not needed; exact grids) --------------
__global__ void k_cvt(const float* __restrict__ src, __bf16* __restrict__ dst, int n)
{
    int i = blockIdx.x * blockDim.x + threadIdx.x;
    if (i < n) dst[i] = f2bf(src[i]);
}

// --- build fused LSTM weight [W_ih || W_hh] (bf16) and bias sum -------------
__global__ void k_prep_lstmw(const float* __restrict__ W_ih, const float* __restrict__ W_hh,
                             const float* __restrict__ b_ih, const float* __restrict__ b_hh,
                             __bf16* __restrict__ Wcat, float* __restrict__ bcat)
{
    const int n = blockIdx.x;      // 512
    const int k = threadIdx.x;     // 256
    const float w = (k < E_) ? W_ih[(size_t)n * E_ + k]
                             : W_hh[(size_t)n * H_ + (k - E_)];
    Wcat[(size_t)n * KCAT + k] = f2bf(w);
    if (k == 0) bcat[n] = b_ih[n] + b_hh[n];
}

// --- state init: xh=[x||h] bf16, c, mask (step-0 update folded in), prob0 ---
__global__ void k_init(const float* __restrict__ dec, const float* __restrict__ h0,
                       const float* __restrict__ c0,
                       __bf16* __restrict__ xh, float* __restrict__ c,
                       unsigned char* __restrict__ mask, float* __restrict__ out0)
{
    const int b = blockIdx.x, t = threadIdx.x;   // 2048 x 128
    xh[(size_t)b * KCAT + t]      = f2bf(dec[(size_t)b * E_ + t]);
    xh[(size_t)b * KCAT + E_ + t] = f2bf(h0[(size_t)b * H_ + t]);
    c[(size_t)b * H_ + t] = c0[(size_t)b * H_ + t];
    if (t < L_) {
        mask[(size_t)b * L_ + t] = (t == 0) ? 1 : 0;  // idx0=0 -> mask[:,0]=True
        out0[(size_t)b * L_ + t] = (t == 0) ? 1.0f : 0.0f;
    }
}

// --- LSTM elementwise: gates -> c_new, h_new (h_new stored bf16 into xh) ----
__global__ void k_lstm(const float* __restrict__ gates, float* __restrict__ c,
                       __bf16* __restrict__ xh)
{
    const int b = blockIdx.x, h = threadIdx.x;   // 2048 x 128
    const float* g = gates + (size_t)b * G4_;
    const float iv = g[h], fv = g[H_ + h], gv = g[2 * H_ + h], ov = g[3 * H_ + h];
    const float si = 1.0f / (1.0f + __expf(-iv));
    const float sf = 1.0f / (1.0f + __expf(-fv));
    const float so = 1.0f / (1.0f + __expf(-ov));
    const float cn = sf * c[(size_t)b * H_ + h] + si * tanhf(gv);
    const float hn = so * tanhf(cn);
    c[(size_t)b * H_ + h] = cn;
    xh[(size_t)b * KCAT + E_ + h] = f2bf(hn);
}

// --- glimpse attention: ug -> softmax -> g_l (bf16) -------------------------
__global__ void k_attn_g(const float* __restrict__ qg, const __bf16* __restrict__ eg,
                         const float* __restrict__ g_v, const unsigned char* __restrict__ mask,
                         __bf16* __restrict__ gl)
{
    const int b = blockIdx.x, l = threadIdx.x;   // 2048 x 64
    __shared__ float sq[H_], sv[H_], su[L_], sp[L_];
    sq[l] = qg[(size_t)b * H_ + l];       sq[l + 64] = qg[(size_t)b * H_ + 64 + l];
    sv[l] = g_v[l];                       sv[l + 64] = g_v[64 + l];
    __syncthreads();

    const __bf16* eb = eg + (size_t)b * H_ * L_;
    float u = 0.0f;
    for (int h = 0; h < H_; ++h) u += sv[h] * tanhf(sq[h] + bf2f(eb[h * L_ + l]));
    if (mask[(size_t)b * L_ + l]) u = NEGC;
    su[l] = u;
    __syncthreads();
    float mx = su[0];
    for (int j = 1; j < L_; ++j) mx = fmaxf(mx, su[j]);
    const float e = __expf(u - mx);
    sp[l] = e;
    __syncthreads();
    float s = 0.0f;
    for (int j = 0; j < L_; ++j) s += sp[j];
    __syncthreads();
    sp[l] = e / s;
    __syncthreads();
    // g_l[h] = sum_l e_g[b,h,l] * p[l]
#pragma unroll
    for (int hh = 0; hh < 2; ++hh) {
        const int h = l + hh * 64;
        float acc = 0.0f;
        for (int j = 0; j < L_; ++j) acc += bf2f(eb[h * L_ + j]) * sp[j];
        gl[(size_t)b * H_ + h] = f2bf(acc);
    }
}

// --- pointer attention: probs out, argmax, mask update, gather next x -------
__global__ void k_attn_p(const float* __restrict__ qp, const __bf16* __restrict__ ep,
                         const float* __restrict__ p_v, unsigned char* __restrict__ mask,
                         const float* __restrict__ emb,  // (L,B,E) f32
                         __bf16* __restrict__ xh,
                         float* __restrict__ probs)      // pre-offset step slice
{
    const int b = blockIdx.x, l = threadIdx.x;   // 2048 x 64
    __shared__ float sq[H_], sv[H_], su[L_], sp[L_];
    sq[l] = qp[(size_t)b * H_ + l];       sq[l + 64] = qp[(size_t)b * H_ + 64 + l];
    sv[l] = p_v[l];                       sv[l + 64] = p_v[64 + l];
    __syncthreads();

    const __bf16* eb = ep + (size_t)b * H_ * L_;
    float u = 0.0f;
    for (int h = 0; h < H_; ++h) u += sv[h] * tanhf(sq[h] + bf2f(eb[h * L_ + l]));
    u = CTANH * tanhf(u);
    if (mask[(size_t)b * L_ + l]) u = NEGC;
    su[l] = u;
    __syncthreads();
    float mx = su[0]; int idx = 0;                 // first-max (matches argmax)
    for (int j = 1; j < L_; ++j) { const float v = su[j]; if (v > mx) { mx = v; idx = j; } }
    const float e = __expf(u - mx);
    sp[l] = e;
    __syncthreads();
    float s = 0.0f;
    for (int j = 0; j < L_; ++j) s += sp[j];
    probs[(size_t)b * L_ + l] = e / s;

    if (l == 0) {   // mask.at[b,idx]=True; then mask[:,0]=False if idx!=0
        if (idx != 0) { mask[(size_t)b * L_ + idx] = 1; mask[(size_t)b * L_] = 0; }
        else          { mask[(size_t)b * L_] = 1; }
    }
    const float* xs = emb + ((size_t)idx * B_ + b) * E_;
    xh[(size_t)b * KCAT + l]      = f2bf(xs[l]);
    xh[(size_t)b * KCAT + 64 + l] = f2bf(xs[64 + l]);
}

// ---------------------------------------------------------------------------
extern "C" void kernel_launch(void* const* d_in, const int* in_sizes, int n_in,
                              void* d_out, int out_size, void* d_ws, size_t ws_size,
                              hipStream_t stream)
{
    (void)in_sizes; (void)n_in; (void)out_size; (void)ws_size;
    const float* dec    = (const float*)d_in[0];
    const float* emb    = (const float*)d_in[1];
    const float* ctx    = (const float*)d_in[2];
    const float* h0     = (const float*)d_in[3];
    const float* c0     = (const float*)d_in[4];
    const float* W_ih   = (const float*)d_in[5];
    const float* W_hh   = (const float*)d_in[6];
    const float* b_ih   = (const float*)d_in[7];
    const float* b_hh   = (const float*)d_in[8];
    const float* g_Wq   = (const float*)d_in[9];
    const float* g_bq   = (const float*)d_in[10];
    const float* g_Wref = (const float*)d_in[11];
    const float* g_bref = (const float*)d_in[12];
    const float* g_v    = (const float*)d_in[13];
    const float* p_Wq   = (const float*)d_in[14];
    const float* p_bq   = (const float*)d_in[15];
    const float* p_Wref = (const float*)d_in[16];
    const float* p_bref = (const float*)d_in[17];
    const float* p_v    = (const float*)d_in[18];
    float* out = (float*)d_out;

    // workspace carve-up (~105 MB total)
    char* base = (char*)d_ws;
    size_t off = 0;
    auto alloc = [&](size_t bytes) -> void* {
        void* p = base + off;
        off = (off + bytes + 255) & ~(size_t)255;
        return p;
    };
    __bf16* xh    = (__bf16*)alloc((size_t)B_ * KCAT * 2);       // [x || h] bf16
    float*  cbuf  = (float*) alloc((size_t)B_ * H_ * 4);
    float*  gates = (float*) alloc((size_t)B_ * G4_ * 4);
    float*  qbuf  = (float*) alloc((size_t)B_ * H_ * 4);         // qg / qp
    __bf16* glb   = (__bf16*)alloc((size_t)B_ * H_ * 2);         // glimpse
    unsigned char* mask = (unsigned char*)alloc((size_t)B_ * L_);
    __bf16* e_g   = (__bf16*)alloc((size_t)B_ * H_ * L_ * 2);    // 32 MB, L2-resident
    __bf16* e_p   = (__bf16*)alloc((size_t)B_ * H_ * L_ * 2);    // 32 MB, L2-resident
    __bf16* ctxb  = (__bf16*)alloc((size_t)L_ * B_ * H_ * 2);
    __bf16* Wcat  = (__bf16*)alloc((size_t)G4_ * KCAT * 2);
    float*  bcat  = (float*) alloc((size_t)G4_ * 4);
    __bf16* gWq   = (__bf16*)alloc((size_t)H_ * H_ * 2);
    __bf16* pWq   = (__bf16*)alloc((size_t)H_ * H_ * 2);
    __bf16* gWr   = (__bf16*)alloc((size_t)H_ * H_ * 2);
    __bf16* pWr   = (__bf16*)alloc((size_t)H_ * H_ * 2);

    // one-time prep
    k_prep_lstmw<<<G4_, KCAT, 0, stream>>>(W_ih, W_hh, b_ih, b_hh, Wcat, bcat);
    k_cvt<<<(H_ * H_) / 256, 256, 0, stream>>>(g_Wq, gWq, H_ * H_);
    k_cvt<<<(H_ * H_) / 256, 256, 0, stream>>>(p_Wq, pWq, H_ * H_);
    k_cvt<<<(H_ * H_) / 256, 256, 0, stream>>>(g_Wref, gWr, H_ * H_);
    k_cvt<<<(H_ * H_) / 256, 256, 0, stream>>>(p_Wref, pWr, H_ * H_);
    k_cvt<<<(L_ * B_ * H_) / 256, 256, 0, stream>>>(ctx, ctxb, L_ * B_ * H_);

    // e_g / e_p precompute: tiles = (L*B/16)*(H/16) = 65536, 8 waves/block
    k_eref<<<65536 / 8, 256, 0, stream>>>(ctxb, gWr, g_bref, e_g);
    k_eref<<<65536 / 8, 256, 0, stream>>>(ctxb, pWr, p_bref, e_p);

    // state init + prob0 slice
    k_init<<<B_, H_, 0, stream>>>(dec, h0, c0, xh, cbuf, mask, out);

    // sequential decode
    for (int s = 0; s < STEPS; ++s) {
        // gates = [x||h] @ Wcat^T + bcat   (M=2048,N=512,K=256) : 4096 tiles
        k_gemm_bias<<<(B_ / 16) * (G4_ / 16) / 8, 256, 0, stream>>>(
            xh, KCAT, Wcat, KCAT, bcat, gates, G4_, G4_ / 16);
        k_lstm<<<B_, H_, 0, stream>>>(gates, cbuf, xh);
        // qg = h_new @ g_Wq^T + g_bq       (M=2048,N=128,K=128) : 1024 tiles
        k_gemm_bias<<<(B_ / 16) * (H_ / 16) / 8, 256, 0, stream>>>(
            xh + E_, KCAT, gWq, H_, g_bq, qbuf, H_, H_ / 16);
        k_attn_g<<<B_, L_, 0, stream>>>(qbuf, e_g, g_v, mask, glb);
        // qp = g_l @ p_Wq^T + p_bq
        k_gemm_bias<<<(B_ / 16) * (H_ / 16) / 8, 256, 0, stream>>>(
            glb, H_, pWq, H_, p_bq, qbuf, H_, H_ / 16);
        k_attn_p<<<B_, L_, 0, stream>>>(qbuf, e_p, p_v, mask, emb, xh,
                                        out + (size_t)(s + 1) * B_ * L_);
    }
}
